// Linear_9491877724571
// MI455X (gfx1250) — 8-bit matrix operand order repaired and checked by reading, compiled, not run on hardware
//
#include <hip/hip_runtime.h>
#include <hip/hip_bf16.h>
#include <stdint.h>


typedef int  v8i  __attribute__((ext_vector_type(8)));
typedef int  v4i  __attribute__((ext_vector_type(4)));
typedef int  v2i  __attribute__((ext_vector_type(2)));
typedef unsigned int u32x4 __attribute__((ext_vector_type(4)));
typedef int  i32x4 __attribute__((ext_vector_type(4)));
typedef int  i32x8 __attribute__((ext_vector_type(8)));

#define T_DIM   16384
#define N_DIM   1024
#define K_DIM   1024
#define EPSF    1e-5f

#define LDS_ROW_STRIDE 80
#define TILE_ROWS      128
#define TILE_KB        64

#define QX_OFF  0
#define QW_OFF  (16777216)
#define AS_OFF  (16777216 + 1048576)
#define WS_OFF  (AS_OFF + 65536)

__global__ void __launch_bounds__(1) init_kernel(float* wsum) {
    *wsum = 0.0f;
}

__global__ void __launch_bounds__(256) act_quant_kernel(
    const float* __restrict__ x, int8_t* __restrict__ qx,
    float* __restrict__ act_scale)
{
    const int lane = threadIdx.x & 31;
    const int wave = threadIdx.x >> 5;
    const int row  = blockIdx.x * 8 + wave;

    const float* xr = x + (size_t)row * K_DIM;
    float4 v[8];
    float amax = 0.0f;
#pragma unroll
    for (int i = 0; i < 8; ++i) {
        v[i] = *(const float4*)(xr + i * 128 + lane * 4);
        amax = fmaxf(amax, fmaxf(fmaxf(fabsf(v[i].x), fabsf(v[i].y)),
                                 fmaxf(fabsf(v[i].z), fabsf(v[i].w))));
    }
#pragma unroll
    for (int off = 16; off >= 1; off >>= 1)
        amax = fmaxf(amax, __shfl_xor(amax, off, 32));

    const float scale = 127.0f / fmaxf(amax, EPSF);
    if (lane == 0) act_scale[row] = scale;

    int8_t* qr = qx + (size_t)row * K_DIM;
#pragma unroll
    for (int i = 0; i < 8; ++i) {
        char4 q;
        int qi;
        qi = (int)rintf(v[i].x * scale); qi = qi < -128 ? -128 : (qi > 127 ? 127 : qi); q.x = (char)qi;
        qi = (int)rintf(v[i].y * scale); qi = qi < -128 ? -128 : (qi > 127 ? 127 : qi); q.y = (char)qi;
        qi = (int)rintf(v[i].z * scale); qi = qi < -128 ? -128 : (qi > 127 ? 127 : qi); q.z = (char)qi;
        qi = (int)rintf(v[i].w * scale); qi = qi < -128 ? -128 : (qi > 127 ? 127 : qi); q.w = (char)qi;
        *(char4*)(qr + i * 128 + lane * 4) = q;
    }
}

__global__ void __launch_bounds__(256) wsum_kernel(
    const float* __restrict__ w, float* __restrict__ wsum)
{
    const int tid = blockIdx.x * 256 + threadIdx.x;
    float s = 0.0f;
#pragma unroll
    for (int i = 0; i < 16; ++i)
        s += fabsf(w[tid + i * 65536]);
#pragma unroll
    for (int off = 16; off >= 1; off >>= 1)
        s += __shfl_xor(s, off, 32);
    if ((threadIdx.x & 31) == 0)
        atomicAdd(wsum, s);
}

__global__ void __launch_bounds__(256) wquant_kernel(
    const float* __restrict__ w, int8_t* __restrict__ qw,
    const float* __restrict__ wsum)
{
    const float mean = *wsum * (1.0f / (float)(N_DIM * K_DIM));
    const float ws   = 1.0f / fmaxf(mean, EPSF);
    const int tid = blockIdx.x * 256 + threadIdx.x;
    const float4 v = *(const float4*)(w + (size_t)tid * 4);
    char4 q;
    int qi;
    qi = (int)rintf(v.x * ws); qi = qi < -1 ? -1 : (qi > 1 ? 1 : qi); q.x = (char)qi;
    qi = (int)rintf(v.y * ws); qi = qi < -1 ? -1 : (qi > 1 ? 1 : qi); q.y = (char)qi;
    qi = (int)rintf(v.z * ws); qi = qi < -1 ? -1 : (qi > 1 ? 1 : qi); q.z = (char)qi;
    qi = (int)rintf(v.w * ws); qi = qi < -1 ? -1 : (qi > 1 ? 1 : qi); q.w = (char)qi;
    *(char4*)(qw + (size_t)tid * 4) = q;
}

#if __has_builtin(__builtin_amdgcn_tensor_load_to_lds)
#define USE_TDM 1
#else
#define USE_TDM 0
#endif

#if USE_TDM
__device__ __forceinline__ void tdm_load_tile_2d(
    const int8_t* gptr, uint32_t lds_addr, uint32_t tensor_rows)
{
    const uint64_t ga = (uint64_t)(uintptr_t)gptr;

    u32x4 g0;
    g0[0] = 1u;
    g0[1] = lds_addr;
    g0[2] = (uint32_t)ga;
    g0[3] = (uint32_t)((ga >> 32) & 0x01FFFFFFu)
          | (2u << 30);

    i32x8 g1;
    g1[0] = (1 << 20)
          | (3 << 22)
          | (3 << 25);
    g1[1] = (int)((uint32_t)(K_DIM & 0xFFFF) << 16);
    g1[2] = (int)((K_DIM >> 16) & 0xFFFF)
          | (int)((tensor_rows & 0xFFFFu) << 16);
    g1[3] = (int)((tensor_rows >> 16) & 0xFFFFu)
          | (TILE_KB << 16);
    g1[4] = TILE_ROWS;
    g1[5] = K_DIM;
    g1[6] = 0;
    g1[7] = 0;

    i32x4 g2; g2[0] = 0; g2[1] = 0; g2[2] = 0; g2[3] = 0;
    i32x4 g3; g3[0] = 0; g3[1] = 0; g3[2] = 0; g3[3] = 0;

#if __clang_major__ >= 23
    i32x8 g4; g4[0]=0; g4[1]=0; g4[2]=0; g4[3]=0; g4[4]=0; g4[5]=0; g4[6]=0; g4[7]=0;
    __builtin_amdgcn_tensor_load_to_lds(g0, g1, g2, g3, g4, 0);
#else
    __builtin_amdgcn_tensor_load_to_lds(g0, g1, g2, g3, 0);
#endif
}

__device__ __forceinline__ uint32_t lds_byte_off(const void* p) {
    return (uint32_t)(uintptr_t)p;
}
#endif

__global__ void __launch_bounds__(256) gemm_iu8_kernel(
    const int8_t* __restrict__ qx, const int8_t* __restrict__ qw,
    const float* __restrict__ act_scale, const float* __restrict__ wsum,
    const float* __restrict__ bias, float* __restrict__ out)
{
    const int lane   = threadIdx.x & 31;
    const int wave   = threadIdx.x >> 5;
    const int laneHi = lane >> 4;
    const int laneLo = lane & 15;

    const int blockRow0 = blockIdx.y * 128;
    const int blockCol0 = blockIdx.x * 128;
    const int wr = wave & 1;
    const int wc = wave >> 1;

    const int aK = laneHi * 16;
    const int bK = laneHi * 16;

    v8i acc[4][2] = {};

#if USE_TDM
    __shared__ __align__(16) int8_t lds_a[2][TILE_ROWS * LDS_ROW_STRIDE];
    __shared__ __align__(16) int8_t lds_b[2][TILE_ROWS * LDS_ROW_STRIDE];

    if (wave == 0)
        tdm_load_tile_2d(qx + (size_t)blockRow0 * K_DIM, lds_byte_off(&lds_a[0][0]), T_DIM);
    else if (wave == 1)
        tdm_load_tile_2d(qw + (size_t)blockCol0 * K_DIM, lds_byte_off(&lds_b[0][0]), N_DIM);
    if (wave < 2) __builtin_amdgcn_s_wait_tensorcnt(0);
    __syncthreads();

    for (int s = 0; s < K_DIM / 64; ++s) {
        const int buf = s & 1;

        if (s + 1 < K_DIM / 64) {
            const int k1 = (s + 1) * 64;
            if (wave == 0)
                tdm_load_tile_2d(qx + (size_t)blockRow0 * K_DIM + k1,
                                 lds_byte_off(&lds_a[buf ^ 1][0]), T_DIM);
            else if (wave == 1)
                tdm_load_tile_2d(qw + (size_t)blockCol0 * K_DIM + k1,
                                 lds_byte_off(&lds_b[buf ^ 1][0]), N_DIM);
        }

        v8i a[4];
#pragma unroll
        for (int i = 0; i < 4; ++i) {
            const int8_t* ap = &lds_a[buf][(wr * 64 + i * 16 + laneLo) * LDS_ROW_STRIDE + aK];
            v2i t0 = *(const v2i*)(ap);
            v2i t1 = *(const v2i*)(ap + 8);
            v2i t2 = *(const v2i*)(ap + 32);
            v2i t3 = *(const v2i*)(ap + 40);
            v8i f;
            f[0] = t0[0]; f[1] = t0[1];
            f[2] = t1[0]; f[3] = t1[1];
            f[4] = t2[0]; f[5] = t2[1];
            f[6] = t3[0]; f[7] = t3[1];
            a[i] = f;
        }
        v8i b[2];
#pragma unroll
        for (int j = 0; j < 2; ++j) {
            const int8_t* bp = &lds_b[buf][(wc * 32 + j * 16 + laneLo) * LDS_ROW_STRIDE + bK];
            v4i u0 = *(const v4i*)(bp);
            v4i u1 = *(const v4i*)(bp + 32);
            v8i f;
            f[0] = u0[0]; f[1] = u0[1]; f[2] = u0[2]; f[3] = u0[3];
            f[4] = u1[0]; f[5] = u1[1]; f[6] = u1[2]; f[7] = u1[3];
            b[j] = f;
        }

#pragma unroll
        for (int i = 0; i < 4; ++i)
#pragma unroll
            for (int j = 0; j < 2; ++j)
                acc[i][j] = __builtin_amdgcn_wmma_i32_16x16x64_iu8(
                     true, a[i],  true, b[j],
                    acc[i][j],  false,  false);

        if (s + 1 < K_DIM / 64) {
            if (wave < 2) __builtin_amdgcn_s_wait_tensorcnt(0);
            __syncthreads();
        }
    }
#else
    const int row_base = blockRow0 + wr * 64;
    const int col_base = blockCol0 + wc * 32;
    for (int k0 = 0; k0 < K_DIM; k0 += 64) {
        v8i a[4];
#pragma unroll
        for (int i = 0; i < 4; ++i) {
            const int8_t* ap = qx + (size_t)(row_base + i * 16 + laneLo) * K_DIM + k0 + aK;
            v2i t0 = *(const v2i*)(ap);
            v2i t1 = *(const v2i*)(ap + 8);
            v2i t2 = *(const v2i*)(ap + 32);
            v2i t3 = *(const v2i*)(ap + 40);
            v8i f;
            f[0] = t0[0]; f[1] = t0[1]; f[2] = t1[0]; f[3] = t1[1];
            f[4] = t2[0]; f[5] = t2[1]; f[6] = t3[0]; f[7] = t3[1];
            a[i] = f;
        }
        v8i b[2];
#pragma unroll
        for (int j = 0; j < 2; ++j) {
            const int8_t* bp = qw + (size_t)(col_base + j * 16 + laneLo) * K_DIM + k0 + bK;
            v4i u0 = *(const v4i*)(bp);
            v4i u1 = *(const v4i*)(bp + 32);
            v8i f;
            f[0] = u0[0]; f[1] = u0[1]; f[2] = u0[2]; f[3] = u0[3];
            f[4] = u1[0]; f[5] = u1[1]; f[6] = u1[2]; f[7] = u1[3];
            b[j] = f;
        }
#pragma unroll
        for (int i = 0; i < 4; ++i)
#pragma unroll
            for (int j = 0; j < 2; ++j)
                acc[i][j] = __builtin_amdgcn_wmma_i32_16x16x64_iu8(
                    true, a[i], true, b[j], acc[i][j], false, false);
    }
#endif

    const float mc = fmaxf(*wsum * (1.0f / (float)(N_DIM * K_DIM)), EPSF);
    const int row_out = blockRow0 + wr * 64;
    const int col_out = blockCol0 + wc * 32;
#pragma unroll
    for (int i = 0; i < 4; ++i) {
#pragma unroll
        for (int r = 0; r < 8; ++r) {
            const int t = row_out + i * 16 + laneHi * 8 + r;
            const float factor = mc / act_scale[t];
#pragma unroll
            for (int j = 0; j < 2; ++j) {
                const int n = col_out + j * 16 + laneLo;
                out[(size_t)t * N_DIM + n] =
                    (float)acc[i][j][r] * factor + bias[n];
            }
        }
    }
}

extern "C" void kernel_launch(void* const* d_in, const int* in_sizes, int n_in,
                              void* d_out, int out_size, void* d_ws, size_t ws_size,
                              hipStream_t stream)
{
    const float* x    = (const float*)d_in[0];
    const float* w    = (const float*)d_in[1];
    const float* bias = (const float*)d_in[2];
    float* out = (float*)d_out;

    int8_t* qx        = (int8_t*)d_ws + QX_OFF;
    int8_t* qw        = (int8_t*)d_ws + QW_OFF;
    float*  act_scale = (float*)((char*)d_ws + AS_OFF);
    float*  wsum      = (float*)((char*)d_ws + WS_OFF);

    init_kernel<<<1, 1, 0, stream>>>(wsum);
    act_quant_kernel<<<T_DIM / 8, 256, 0, stream>>>(x, qx, act_scale);
    wsum_kernel<<<256, 256, 0, stream>>>(w, wsum);
    wquant_kernel<<<(N_DIM * K_DIM) / (4 * 256), 256, 0, stream>>>(w, qw, wsum);

    dim3 grid(N_DIM / 128, T_DIM / 128);
    gemm_iu8_kernel<<<grid, 256, 0, stream>>>(qx, qw, act_scale, wsum, bias, out);
}
